// UniGCNLayer_84954453115307
// MI455X (gfx1250) — matrix operand order repaired and checked by reading, not compiled, not run on hardware
//
#include <hip/hip_runtime.h>
#include <hip/hip_bf16.h>

// Problem constants (fixed by the harness / reference).
#define N_NODES 100000
#define N_EDGES 20000
#define NNZ     1600000
#define IN_CH   128
#define HID_CH  256

typedef __bf16 v16bf __attribute__((ext_vector_type(16)));
typedef float  v8f   __attribute__((ext_vector_type(8)));

// ---------------------------------------------------------------------------
// Zero-fill (float4 grid-stride).
// ---------------------------------------------------------------------------
__global__ void zero_f4(float4* __restrict__ p, size_t n4) {
    size_t i      = (size_t)blockIdx.x * blockDim.x + threadIdx.x;
    size_t stride = (size_t)gridDim.x * blockDim.x;
    float4 z = make_float4(0.f, 0.f, 0.f, 0.f);
    for (; i < n4; i += stride) p[i] = z;
}

// ---------------------------------------------------------------------------
// Scatter level 1:  y[edge_idx[i]] += x0[node_idx[i]]   (128 channels)
// One wave per nnz; each lane handles 4 channels (float4 gather + 4 fp atomics).
// ---------------------------------------------------------------------------
__global__ void scatter_n2e(const float* __restrict__ x0,
                            const int*   __restrict__ nidx,
                            const int*   __restrict__ eidx,
                            float*       __restrict__ y) {
    const int i    = (int)(((size_t)blockIdx.x * blockDim.x + threadIdx.x) >> 5);
    const int lane = threadIdx.x & 31;
    if (i >= NNZ) return;
    const int node = nidx[i];
    const int edge = eidx[i];
    const float4 v = *(const float4*)(x0 + (size_t)node * IN_CH + lane * 4);
    float* dst = y + (size_t)edge * IN_CH + lane * 4;
    unsafeAtomicAdd(dst + 0, v.x);
    unsafeAtomicAdd(dst + 1, v.y);
    unsafeAtomicAdd(dst + 2, v.z);
    unsafeAtomicAdd(dst + 3, v.w);
}

// ---------------------------------------------------------------------------
// Scatter level 2:  out[node_idx[i]] += m[edge_idx[i]]  (256 channels)
// One wave per nnz; each lane handles 8 channels.
// ---------------------------------------------------------------------------
__global__ void scatter_e2n(const float* __restrict__ m,
                            const int*   __restrict__ eidx,
                            const int*   __restrict__ nidx,
                            float*       __restrict__ out) {
    const int i    = (int)(((size_t)blockIdx.x * blockDim.x + threadIdx.x) >> 5);
    const int lane = threadIdx.x & 31;
    if (i >= NNZ) return;
    const int edge = eidx[i];
    const int node = nidx[i];
    const float4* src = (const float4*)(m + (size_t)edge * HID_CH + lane * 8);
    const float4 v0 = src[0];
    const float4 v1 = src[1];
    float* dst = out + (size_t)node * HID_CH + lane * 8;
    unsafeAtomicAdd(dst + 0, v0.x);
    unsafeAtomicAdd(dst + 1, v0.y);
    unsafeAtomicAdd(dst + 2, v0.z);
    unsafeAtomicAdd(dst + 3, v0.w);
    unsafeAtomicAdd(dst + 4, v1.x);
    unsafeAtomicAdd(dst + 5, v1.y);
    unsafeAtomicAdd(dst + 6, v1.z);
    unsafeAtomicAdd(dst + 7, v1.w);
}

// ---------------------------------------------------------------------------
// GEMM C[M,N] = A[M,K] * B[K,N]  (fp32 in/out) via bf16x3 WMMA emulation:
//   a = ahi + alo, b = bhi + blo,  a*b ~= ahi*bhi + ahi*blo + alo*bhi
// accumulated in f32 with v_wmma_f32_16x16x32_bf16.
// One wave per 16x16 output tile. M % 16 == 0, N % 16 == 0, K % 32 == 0.
//
// VGPR fragment layouts per CDNA5 ISA 7.12.2:
//   A (16x32 bf16): lane L holds row M=(L&15); elements 0..7 -> K = k0+q8+e,
//                   elements 8..15 -> K = k0+q8+16+(e-8), q8 = (L>=16)*8.
//   B (32x16 bf16): lane L holds col N=(L&15); same K mapping as A:
//                   e<8 -> K = k0+q8+e, e>=8 -> K = k0+q8+16+(e-8).
//   C/D (16x16 f32): VGPR j: lanes 0-15 -> (M=j, N=lane),
//                            lanes 16-31 -> (M=j+8, N=lane-16).
// ---------------------------------------------------------------------------
__global__ void gemm_bf16x3(const float* __restrict__ A,
                            const float* __restrict__ B,
                            float*       __restrict__ C,
                            int M, int K, int N) {
    const int lane   = threadIdx.x & 31;
    const int waveIB = threadIdx.x >> 5;
    const int tilesN = N >> 4;
    const int tilesM = M >> 4;
    const int tile   = blockIdx.x * (blockDim.x >> 5) + waveIB;
    if (tile >= tilesM * tilesN) return;   // wave-uniform: EXEC stays all-ones

    const int tm  = tile / tilesN;
    const int tn  = tile - tm * tilesN;
    const int l15 = lane & 15;
    const int hh  = lane >> 4;             // half selector: 0 or 1

    const float* Arow = A + (size_t)(tm * 16 + l15) * K + hh * 8;   // + q8
    const float* Bcol = B + (size_t)(hh * 8) * N + tn * 16 + l15;   // + q8 rows

    v8f acc = {0.f, 0.f, 0.f, 0.f, 0.f, 0.f, 0.f, 0.f};

    for (int kk = 0; kk < K; kk += 32) {
        // ---- A fragment: 16 contiguous-ish f32 -> bf16 hi/lo split ----
        const float4* pa0 = (const float4*)(Arow + kk);       // K = kk+q8+0..7
        const float4* pa1 = (const float4*)(Arow + kk + 16);  // K = kk+q8+16..23
        const float4 a0 = pa0[0], a1 = pa0[1];
        const float4 a2 = pa1[0], a3 = pa1[1];
        float af[16] = {a0.x, a0.y, a0.z, a0.w, a1.x, a1.y, a1.z, a1.w,
                        a2.x, a2.y, a2.z, a2.w, a3.x, a3.y, a3.z, a3.w};
        v16bf ahi, alo;
#pragma unroll
        for (int e = 0; e < 16; ++e) {
            __bf16 h = (__bf16)af[e];
            ahi[e] = h;
            alo[e] = (__bf16)(af[e] - (float)h);
        }

        // ---- B fragment: strided column reads (W is L2/WGP$ resident) ----
        const float* Bp = Bcol + (size_t)kk * N;
        v16bf bhi, blo;
#pragma unroll
        for (int e = 0; e < 16; ++e) {
            float f = Bp[(size_t)(e + (e & 8)) * N];
            __bf16 h = (__bf16)f;
            bhi[e] = h;
            blo[e] = (__bf16)(f - (float)h);
        }

        // ---- 3-term bf16 emulated fp32 MMA, f32 accumulate ----
        acc = __builtin_amdgcn_wmma_f32_16x16x32_bf16(
                  false, ahi, false, bhi, (short)0, acc, false, false);
        acc = __builtin_amdgcn_wmma_f32_16x16x32_bf16(
                  false, ahi, false, blo, (short)0, acc, false, false);
        acc = __builtin_amdgcn_wmma_f32_16x16x32_bf16(
                  false, alo, false, bhi, (short)0, acc, false, false);
    }

    // ---- Store C fragment ----
    float* Crow = C + (size_t)(tm * 16 + hh * 8) * N + tn * 16 + l15;
#pragma unroll
    for (int j = 0; j < 8; ++j) Crow[(size_t)j * N] = acc[j];
}

// ---------------------------------------------------------------------------
// Launch: zero y -> scatter1 -> x1 = y*W1 -> m = x1*W2 -> zero out -> scatter2
// ---------------------------------------------------------------------------
extern "C" void kernel_launch(void* const* d_in, const int* in_sizes, int n_in,
                              void* d_out, int out_size, void* d_ws, size_t ws_size,
                              hipStream_t stream) {
    const float* x0   = (const float*)d_in[0];
    const int*   nidx = (const int*)d_in[1];
    const int*   eidx = (const int*)d_in[2];
    // d_in[3] = n_edges (scalar on device) -- compile-time constant N_EDGES
    const float* W1   = (const float*)d_in[4];
    const float* W2   = (const float*)d_in[5];

    float* x0_out = (float*)d_out;                                  // 100000*256
    float* x1_out = (float*)d_out + (size_t)N_NODES * HID_CH;       // 20000*256
    float* y      = (float*)d_ws;                                   // 20000*128
    float* m      = (float*)d_ws;                                   // 20000*256 (y dead by then)

    // 1) y = 0
    zero_f4<<<2048, 256, 0, stream>>>((float4*)y, (size_t)N_EDGES * IN_CH / 4);

    // 2) y[e] += x0[n]  (B1^T gather-scatter, 128 ch)
    scatter_n2e<<<NNZ / 8, 256, 0, stream>>>(x0, nidx, eidx, y);

    // 3) x1 = y @ W1   (20000x128 @ 128x256) -> straight into d_out
    {
        int tiles = (N_EDGES / 16) * (HID_CH / 16);
        gemm_bf16x3<<<(tiles + 7) / 8, 256, 0, stream>>>(y, W1, x1_out,
                                                         N_EDGES, IN_CH, HID_CH);
    }

    // 4) m = x1 @ W2   (20000x256 @ 256x256) -> workspace
    {
        int tiles = (N_EDGES / 16) * (HID_CH / 16);
        gemm_bf16x3<<<(tiles + 7) / 8, 256, 0, stream>>>(x1_out, W2, m,
                                                         N_EDGES, HID_CH, HID_CH);
    }

    // 5) x0_out = 0
    zero_f4<<<4096, 256, 0, stream>>>((float4*)x0_out, (size_t)N_NODES * HID_CH / 4);

    // 6) x0_out[n] += m[e]  (B1 gather-scatter, 256 ch)
    scatter_e2n<<<NNZ / 8, 256, 0, stream>>>(m, eidx, nidx, x0_out);
}
